// SupervisedContrastiveLoss_64759516889387
// MI455X (gfx1250) — matrix operand order repaired and checked by reading, not compiled, not run on hardware
//
#include <hip/hip_runtime.h>

typedef __attribute__((ext_vector_type(16))) _Float16 v16h;
typedef __attribute__((ext_vector_type(8)))  _Float16 v8h;
typedef __attribute__((ext_vector_type(8)))  float    v8f;
typedef __attribute__((ext_vector_type(4)))  int      v4i;

#define B_N        8192
#define D_K        256
#define INV_T      14.285714285714286f   // 1/0.07; also the fixed logsumexp max M0
#define WAVES_WG   8
#define ROWS_WG    (WAVES_WG * 16)       // 128 rows per workgroup
#define COL_CHUNKS 8
#define TILES_CHNK (B_N / 16 / COL_CHUNKS) // 64 16-col tiles per chunk

// CDNA5 async global->LDS path (ASYNCcnt-tracked, §08).  Signature recovered
// from the round-2 diagnostic: (v4i as(1)*, v4i as(3)*, imm offset, imm cpol).
#if defined(__has_builtin)
#  if __has_builtin(__builtin_amdgcn_global_load_async_to_lds_b128) && \
      __has_builtin(__builtin_amdgcn_s_wait_asynccnt)
#    define USE_ASYNC_LDS 1
#  endif
#endif
#ifndef USE_ASYNC_LDS
#  define USE_ASYNC_LDS 0
#endif

#if USE_ASYNC_LDS
typedef __attribute__((address_space(1))) v4i gv4i;   // global int4
typedef __attribute__((address_space(3))) v4i lv4i;   // LDS int4
__device__ __forceinline__ void async_copy_32B(const _Float16* gsrc, _Float16* ldst) {
    gv4i* g = (gv4i*)gsrc;
    lv4i* l = (lv4i*)ldst;
    __builtin_amdgcn_global_load_async_to_lds_b128(g, l, 0, 0);
    __builtin_amdgcn_global_load_async_to_lds_b128(g, l, 16, 0);
}
#endif

// ---------------- workspace layout (floats) ----------------
// [0..63]                 : header (hdr[0]=sum of mean_log_prob_pos, hdr[1]=n_valid)
// [64 .. 64+8191]         : row_s    (sum of exp(s - M0), off-diagonal)
// [.. +8192]              : row_pos  (sum of sim over positives)
// [.. +8192]              : row_npos (count of positives)
// then, 16B-aligned       : feats16  (8192 x 256 _Float16, row-major)
#define HDR_FLOATS (64 + 3 * B_N)

// ---------------------------------------------------------------------------
// 0) zero the accumulators
// ---------------------------------------------------------------------------
__global__ void init_ws_kernel(float* ws) {
    int i = blockIdx.x * 256 + threadIdx.x;
    if (i < HDR_FLOATS) ws[i] = 0.0f;
}

// ---------------------------------------------------------------------------
// 1) L2-normalize rows, emit f16.  One wave (32 lanes) per row, 8 elems/lane.
// ---------------------------------------------------------------------------
__global__ __launch_bounds__(256) void normalize_kernel(const float* __restrict__ x,
                                                        _Float16* __restrict__ y) {
    const int wave = threadIdx.x >> 5;
    const int lane = threadIdx.x & 31;
    const int row  = blockIdx.x * 8 + wave;

    const float* xr = x + (size_t)row * D_K + lane * 8;
    float v[8];
    float ss = 0.0f;
#pragma unroll
    for (int i = 0; i < 8; ++i) { v[i] = xr[i]; ss += v[i] * v[i]; }
#pragma unroll
    for (int m = 16; m >= 1; m >>= 1) ss += __shfl_xor(ss, m, 32);
    const float inv = rsqrtf(ss);

    v8h h;
#pragma unroll
    for (int i = 0; i < 8; ++i) h[i] = (_Float16)(v[i] * inv);
    *(v8h*)(y + (size_t)row * D_K + lane * 8) = h;
}

// ---------------------------------------------------------------------------
// 2) Fused similarity GEMM + contrastive epilogue, software-pipelined B tiles.
// ---------------------------------------------------------------------------
__global__ __launch_bounds__(256) void simloss_kernel(const _Float16* __restrict__ f16,
                                                      const int* __restrict__ conc,
                                                      const int* __restrict__ cls,
                                                      float* __restrict__ row_s,
                                                      float* __restrict__ row_pos,
                                                      float* __restrict__ row_npos) {
#if USE_ASYNC_LDS
    __shared__ _Float16 Bt[2][16 * D_K];   // 2 x 8 KB, double-buffered
#else
    __shared__ _Float16 Bt[1][16 * D_K];   // 8 KB, register-staged pipeline
#endif

    const int lane = threadIdx.x & 31;
    const int wave = threadIdx.x >> 5;
    const int l15  = lane & 15;
    const int hi   = lane >> 4;

    const int wave_row = blockIdx.x * ROWS_WG + wave * 16;

    // ---- A fragments: 16x32 f16 per fragment, per ISA 16-bit A layout:
    // lane l15 = row M; lanes 0-15 hold K {0..7,16..23}, lanes 16-31 hold {8..15,24..31}
    v16h afrag[8];
    {
        const _Float16* ap = f16 + (size_t)(wave_row + l15) * D_K;
#pragma unroll
        for (int f = 0; f < 8; ++f) {
            v8h lo = *(const v8h*)(ap + f * 32 + hi * 8);
            v8h up = *(const v8h*)(ap + f * 32 + 16 + hi * 8);
#pragma unroll
            for (int e = 0; e < 8; ++e) { afrag[f][e] = lo[e]; afrag[f][e + 8] = up[e]; }
        }
    }

    // ---- per-lane row metadata: C/D layout -> VGPR v holds row (hi*8 + v)
    int rown[8], rc[8], rk[8];
#pragma unroll
    for (int v = 0; v < 8; ++v) {
        rown[v] = wave_row + hi * 8 + v;
        rc[v]   = conc[rown[v]];
        rk[v]   = cls[rown[v]];
    }

    float acc_s[8], acc_p[8], acc_n[8];
#pragma unroll
    for (int v = 0; v < 8; ++v) { acc_s[v] = 0.f; acc_p[v] = 0.f; acc_n[v] = 0.f; }

    const int t0 = blockIdx.y * TILES_CHNK;

#if USE_ASYNC_LDS
    // Each thread stages 32 B of the 8 KB tile with 2 async b128 ops (ASYNCcnt).
    const size_t stg = (size_t)threadIdx.x * 16;  // halfs
    async_copy_32B(f16 + (size_t)t0 * 16 * D_K + stg, &Bt[0][stg]);
#else
    v8h r0, r1;
    {
        const v8h* src = (const v8h*)(f16 + (size_t)t0 * 16 * D_K);
        r0 = src[threadIdx.x * 2];
        r1 = src[threadIdx.x * 2 + 1];
    }
#endif

    for (int i = 0; i < TILES_CHNK; ++i) {
        const int t = t0 + i;
#if USE_ASYNC_LDS
        const int cur = i & 1;
        if (i + 1 < TILES_CHNK) {   // prefetch tile t+1 into the other buffer
            async_copy_32B(f16 + (size_t)(t + 1) * 16 * D_K + stg, &Bt[cur ^ 1][stg]);
            __builtin_amdgcn_s_wait_asynccnt(2);  // buf[cur] done; prefetch in flight
        } else {
            __builtin_amdgcn_s_wait_asynccnt(0);
        }
        __syncthreads();                          // buf[cur] visible to all waves
        const _Float16* Bcur = Bt[cur];
#else
        __syncthreads();                          // prev iteration's LDS reads done
        {
            v8h* dst = (v8h*)Bt[0];
            dst[threadIdx.x * 2]     = r0;
            dst[threadIdx.x * 2 + 1] = r1;
        }
        if (i + 1 < TILES_CHNK) {                 // overlap next fetch with compute
            const v8h* src = (const v8h*)(f16 + (size_t)(t + 1) * 16 * D_K);
            r0 = src[threadIdx.x * 2];
            r1 = src[threadIdx.x * 2 + 1];
        }
        __syncthreads();                          // tile visible to all waves
        const _Float16* Bcur = Bt[0];
#endif

        // ---- 16x16 sim tile: K=256 in 8 WMMA steps of 32
        v8f acc = {};
#pragma unroll
        for (int f = 0; f < 8; ++f) {
            // B layout: lane l15 = col N; lanes 0-15 hold K f*32+{0..7,16..23}, 16-31 hold +{8..15,24..31}
            const _Float16* bp = Bcur + l15 * D_K + f * 32 + hi * 8;
            v8h b0 = *(const v8h*)bp;
            v8h b1 = *(const v8h*)(bp + 16);
            v16h bf;
#pragma unroll
            for (int e = 0; e < 8; ++e) { bf[e] = b0[e]; bf[e + 8] = b1[e]; }
            acc = __builtin_amdgcn_wmma_f32_16x16x32_f16(
                /*neg_a=*/false, afrag[f], /*neg_b=*/false, bf,
                /*c_mod=*/(short)0, acc, /*reuse_a=*/false, /*reuse_b=*/false);
        }

        // ---- epilogue: fixed-max logsumexp partial + positive-mask partials
        const int col = t * 16 + l15;
        const int cc  = conc[col];
        const int ck  = cls[col];
#pragma unroll
        for (int v = 0; v < 8; ++v) {
            const float s   = acc[v] * INV_T;          // sim / temperature
            const bool diag = (rown[v] == col);
            const float ex  = __expf(s - INV_T);       // exp(s - M0), in (0,1]
            acc_s[v] += diag ? 0.0f : ex;              // diag contributes exp(-1e9)=0
            const bool pos = (cc == rc[v]) && (ck == rk[v]) && !diag;
            acc_p[v] += pos ? s : 0.0f;
            acc_n[v] += pos ? 1.0f : 0.0f;
        }

#if USE_ASYNC_LDS
        __syncthreads();   // all waves done reading buf[cur] before it is re-staged
#endif
    }

    // ---- reduce across the 16 lanes of each half (column dimension)
#pragma unroll
    for (int v = 0; v < 8; ++v) {
#pragma unroll
        for (int m = 8; m >= 1; m >>= 1) {
            acc_s[v] += __shfl_xor(acc_s[v], m, 32);
            acc_p[v] += __shfl_xor(acc_p[v], m, 32);
            acc_n[v] += __shfl_xor(acc_n[v], m, 32);
        }
    }
    // lanes 0-7 commit rows hi=0, lanes 16-23 commit rows hi=1 (linear merges)
    if (l15 < 8) {
        const int v = l15;
        atomicAdd(&row_s[rown[v]],    acc_s[v]);
        atomicAdd(&row_pos[rown[v]],  acc_p[v]);
        atomicAdd(&row_npos[rown[v]], acc_n[v]);
    }
}

// ---------------------------------------------------------------------------
// 3) per-row finalize: lse = M0 + log(row_s); mean_log_prob_pos; wave-reduced
//    atomic accumulation of (sum, n_valid) into header.
// ---------------------------------------------------------------------------
__global__ __launch_bounds__(256) void finalize_rows_kernel(const float* __restrict__ row_s,
                                                            const float* __restrict__ row_pos,
                                                            const float* __restrict__ row_npos,
                                                            float* __restrict__ hdr) {
    const int row = blockIdx.x * 256 + threadIdx.x;
    float contrib = 0.0f, valid = 0.0f;
    if (row < B_N) {
        const float np = row_npos[row];
        if (np > 0.0f) {
            const float lse = INV_T + __logf(row_s[row]);
            contrib = (row_pos[row] - np * lse) / (np + 1e-9f);
            valid   = 1.0f;
        }
    }
#pragma unroll
    for (int m = 16; m >= 1; m >>= 1) {
        contrib += __shfl_xor(contrib, m, 32);
        valid   += __shfl_xor(valid,   m, 32);
    }
    if ((threadIdx.x & 31) == 0) {
        atomicAdd(&hdr[0], contrib);
        atomicAdd(&hdr[1], valid);
    }
}

__global__ void finalize_scalar_kernel(const float* __restrict__ hdr,
                                       float* __restrict__ out) {
    if (threadIdx.x == 0) {
        const float nv = hdr[1];
        out[0] = (nv > 0.0f) ? (-hdr[0] / fmaxf(nv, 1.0f)) : 0.0f;
    }
}

// ---------------------------------------------------------------------------
extern "C" void kernel_launch(void* const* d_in, const int* in_sizes, int n_in,
                              void* d_out, int out_size, void* d_ws, size_t ws_size,
                              hipStream_t stream) {
    (void)in_sizes; (void)n_in; (void)out_size; (void)ws_size;

    const float* feats = (const float*)d_in[0];
    const int*   conc  = (const int*)d_in[1];
    const int*   cls   = (const int*)d_in[2];
    float*       out   = (float*)d_out;

    float* ws       = (float*)d_ws;
    float* hdr      = ws;
    float* row_s    = ws + 64;
    float* row_pos  = ws + 64 + B_N;
    float* row_npos = ws + 64 + 2 * B_N;
    _Float16* f16   = (_Float16*)((char*)d_ws + (size_t)HDR_FLOATS * sizeof(float));

    init_ws_kernel<<<(HDR_FLOATS + 255) / 256, 256, 0, stream>>>(ws);
    normalize_kernel<<<B_N / 8, 256, 0, stream>>>(feats, f16);
    dim3 grid(B_N / ROWS_WG, COL_CHUNKS);   // 64 x 8 = 512 workgroups
    simloss_kernel<<<grid, 256, 0, stream>>>(f16, conc, cls, row_s, row_pos, row_npos);
    finalize_rows_kernel<<<B_N / 256, 256, 0, stream>>>(row_s, row_pos, row_npos, hdr);
    finalize_scalar_kernel<<<1, 32, 0, stream>>>(hdr, out);
}
